// DCNv2_SA_27522150432907
// MI455X (gfx1250) — matrix operand order repaired and checked by reading, not compiled, not run on hardware
//
#include <hip/hip_runtime.h>

typedef __attribute__((ext_vector_type(16))) __bf16        v16bf;
typedef __attribute__((ext_vector_type(2)))  __bf16        v2bf;
typedef __attribute__((ext_vector_type(8)))  float         v8f;
typedef __attribute__((ext_vector_type(4)))  unsigned int  uint4v;
typedef __attribute__((ext_vector_type(2)))  unsigned int  uint2v;

#define CIN   256
#define COUT  256
#define HH    96
#define WW    96
#define HWSZ  (HH*WW)              // 9216
#define NB    4
#define NPOS  (NB*HWSZ)            // 36864
#define GRP   8
#define GELEM (32*HWSZ)            // 294912 elements per (b,g)
#define NCHUNK 36

// ---- f32 pair -> packed bf16 (RNE) ---------------------------------------
__device__ __forceinline__ unsigned int pack2bf(float a, float b) {
#if __has_builtin(__builtin_amdgcn_cvt_pk_bf16_f32)
  union { v2bf v; unsigned int u; } r;
  r.v = __builtin_amdgcn_cvt_pk_bf16_f32(a, b);
  return r.u;
#else
  unsigned int ua = __float_as_uint(a);
  unsigned int ub = __float_as_uint(b);
  ua += 0x7FFFu + ((ua >> 16) & 1u);
  ub += 0x7FFFu + ((ub >> 16) & 1u);
  return (ua >> 16) | (ub & 0xFFFF0000u);
#endif
}

// ==========================================================================
// Kernel 0: repack weight[O][Cin][3][3] f32 -> bf16 B-fragments in WMMA
// register order. Linear: ((((tap*8+q)*16+nt)*32)+lane)*8 + j
// ==========================================================================
__global__ __launch_bounds__(256) void k_repack(const float* __restrict__ w,
                                                unsigned int* __restrict__ WB) {
  int idx  = blockIdx.x * 256 + threadIdx.x;      // 294912 total
  int j    = idx & 7;
  int lane = (idx >> 3) & 31;
  int nt   = (idx >> 8) & 15;
  int q    = (idx >> 12) & 7;
  int tap  = idx >> 15;
  int n    = nt * 16 + (lane & 15);
  int h    = lane >> 4;
  int k0   = q * 32 + 16 * h + 2 * j;
  float w0 = w[(n * CIN + k0) * 9 + tap];
  float w1 = w[(n * CIN + k0 + 1) * 9 + tap];
  WB[idx] = pack2bf(w0, w1);
}

// ==========================================================================
// Kernel 1: fused DCNv2 implicit GEMM, bf16 WMMA, M-tile=64, double-buffered
// A tile, 1 barrier per K-step, 4 wmma per wave per step.
// Block = 512 threads (16 waves) : wave owns 16 Couts; sampling role:
// pos_i = (lane)+(wave&1)*32, channel group cgrp = wave>>1 (4 channels).
// ==========================================================================
__global__ __launch_bounds__(512) void k_dcn_wmma(const float* __restrict__ x,
                                                  const float* __restrict__ off,
                                                  const float* __restrict__ msk,
                                                  const unsigned int* __restrict__ WB,
                                                  float* __restrict__ conv) {
  __shared__ __align__(16) unsigned short sA[2][64 * 40];  // 2 x 64 rows x 80B
  __shared__ float sD[COUT][33];

  const int tid  = threadIdx.x;
  const int wave = tid >> 5;
  const int lane = tid & 31;
  const int pos0 = blockIdx.x * 64;
  const int b    = pos0 / HWSZ;          // 64 | 9216 -> uniform b
  const int hw0  = pos0 - b * HWSZ;

  // ---- sampling role ----
  const int pos_i = lane + ((wave & 1) << 5);   // 0..63
  const int cgrp  = wave >> 1;                  // 0..7 (4 channels each)
  const int hwp   = hw0 + pos_i;
  const int yyp   = hwp / WW;
  const int xxp   = hwp - yyp * WW;

  const float* xb    = x + (size_t)b * (CIN * HWSZ) + (size_t)(cgrp * 4) * HWSZ;
  const int    obase = ((b * 18) * HH + yyp) * WW + xxp;
  const int    mbase = ((b * 9)  * HH + yyp) * WW + xxp;

  float w00, w01, w10, w11;
  const float *a00, *a01, *a10, *a11;

  auto computeParams = [&](int tap) {
    int ky = tap / 3, kx = tap - ky * 3;
    float dy = off[obase + (2 * tap) * HWSZ];
    float dx = off[obase + (2 * tap + 1) * HWSZ];
    float mv = msk[mbase + tap * HWSZ];
    float py = (float)(yyp + ky - 1) + dy;
    float px = (float)(xxp + kx - 1) + dx;
    float fy = floorf(py), fx = floorf(px);
    float wy = py - fy,    wx = px - fx;
    int y0 = (int)fy, x0i = (int)fx;
    int y1 = y0 + 1,  x1i = x0i + 1;
    float vy0 = (y0 >= 0 && y0 < HH) ? 1.f : 0.f;
    float vy1 = (y1 >= 0 && y1 < HH) ? 1.f : 0.f;
    float vx0 = (x0i >= 0 && x0i < WW) ? 1.f : 0.f;
    float vx1 = (x1i >= 0 && x1i < WW) ? 1.f : 0.f;
    w00 = (1.f - wy) * (1.f - wx) * mv * vy0 * vx0;
    w01 = (1.f - wy) * wx * mv * vy0 * vx1;
    w10 = wy * (1.f - wx) * mv * vy1 * vx0;
    w11 = wy * wx * mv * vy1 * vx1;
    int y0c = min(max(y0, 0), HH - 1), y1c = min(max(y1, 0), HH - 1);
    int x0c = min(max(x0i, 0), WW - 1), x1c = min(max(x1i, 0), WW - 1);
    a00 = xb + y0c * WW + x0c;  a01 = xb + y0c * WW + x1c;
    a10 = xb + y1c * WW + x0c;  a11 = xb + y1c * WW + x1c;
  };

  // full sample (gather + combine + pack + store) for chunk q into buffer buf
  auto fullSample = [&](int q, int bufDst) {
    float g[16];
#pragma unroll
    for (int k = 0; k < 4; ++k) {
      g[4 * k + 0] = a00[(q * 32 + k) * HWSZ];
      g[4 * k + 1] = a01[(q * 32 + k) * HWSZ];
      g[4 * k + 2] = a10[(q * 32 + k) * HWSZ];
      g[4 * k + 3] = a11[(q * 32 + k) * HWSZ];
    }
    float v0 = w00 * g[0]  + w01 * g[1]  + w10 * g[2]  + w11 * g[3];
    float v1 = w00 * g[4]  + w01 * g[5]  + w10 * g[6]  + w11 * g[7];
    float v2 = w00 * g[8]  + w01 * g[9]  + w10 * g[10] + w11 * g[11];
    float v3 = w00 * g[12] + w01 * g[13] + w10 * g[14] + w11 * g[15];
    uint2v d; d.x = pack2bf(v0, v1); d.y = pack2bf(v2, v3);
    *(uint2v*)&sA[bufDst][pos_i * 40 + cgrp * 4] = d;
  };

  v8f acc0 = {0.f,0.f,0.f,0.f,0.f,0.f,0.f,0.f};
  v8f acc1 = acc0, acc2 = acc0, acc3 = acc0;

  const unsigned int* wb0 = WB + (size_t)((wave * 32 + lane) * 8);
  const int mrow = lane & 15, khalf = lane >> 4;

  // ---- prologue: fill buffer 0 with (tap0, chunk0) ----
  computeParams(0);
  fullSample(0, 0);
  __syncthreads();

#pragma unroll 1
  for (int tap = 0; tap < 9; ++tap) {
#pragma unroll
    for (int q = 0; q < 8; ++q) {
      const int it = tap * 8 + q;
      // 1) B fragments (global, coalesced, register-order) -- issued FIRST so
      //    wmma's loadcnt wait does not cover the gathers issued below.
      union { v16bf v; uint4v u[2]; } Bf;
      Bf.u[0] = *(const uint4v*)(wb0 + (size_t)it * 4096);
      Bf.u[1] = *(const uint4v*)(wb0 + (size_t)it * 4096 + 4);

      // 2) next-chunk gathers (issue only; combine deferred past the wmmas)
      float g[16];
      if (q < 7) {
#pragma unroll
        for (int k = 0; k < 4; ++k) {
          g[4 * k + 0] = a00[((q + 1) * 32 + k) * HWSZ];
          g[4 * k + 1] = a01[((q + 1) * 32 + k) * HWSZ];
          g[4 * k + 2] = a10[((q + 1) * 32 + k) * HWSZ];
          g[4 * k + 3] = a11[((q + 1) * 32 + k) * HWSZ];
        }
      }

      // 3) interleaved A-fragment loads + wmma (each wmma waits only its quad)
      const char* abp = (const char*)sA[q & 1] + mrow * 80 + khalf * 32;
      union { v16bf v; uint4v u[2]; } A;
      A.u[0] = *(const uint4v*)(abp);
      A.u[1] = *(const uint4v*)(abp + 16);
      acc0 = __builtin_amdgcn_wmma_f32_16x16x32_bf16(false, A.v, false, Bf.v,
                                                     (short)0, acc0, false, false);
      A.u[0] = *(const uint4v*)(abp + 1280);
      A.u[1] = *(const uint4v*)(abp + 1296);
      acc1 = __builtin_amdgcn_wmma_f32_16x16x32_bf16(false, A.v, false, Bf.v,
                                                     (short)0, acc1, false, false);
      A.u[0] = *(const uint4v*)(abp + 2560);
      A.u[1] = *(const uint4v*)(abp + 2576);
      acc2 = __builtin_amdgcn_wmma_f32_16x16x32_bf16(false, A.v, false, Bf.v,
                                                     (short)0, acc2, false, false);
      A.u[0] = *(const uint4v*)(abp + 3840);
      A.u[1] = *(const uint4v*)(abp + 3856);
      acc3 = __builtin_amdgcn_wmma_f32_16x16x32_bf16(false, A.v, false, Bf.v,
                                                     (short)0, acc3, false, false);

      // 4) combine + pack + store into the other buffer (after the wmmas)
      if (q < 7) {
        float v0 = w00 * g[0]  + w01 * g[1]  + w10 * g[2]  + w11 * g[3];
        float v1 = w00 * g[4]  + w01 * g[5]  + w10 * g[6]  + w11 * g[7];
        float v2 = w00 * g[8]  + w01 * g[9]  + w10 * g[10] + w11 * g[11];
        float v3 = w00 * g[12] + w01 * g[13] + w10 * g[14] + w11 * g[15];
        uint2v d; d.x = pack2bf(v0, v1); d.y = pack2bf(v2, v3);
        *(uint2v*)&sA[(q + 1) & 1][pos_i * 40 + cgrp * 4] = d;
      } else if (tap < 8) {
        computeParams(tap + 1);
        fullSample(0, 0);          // next tap always starts at buffer 0
      }
      __syncthreads();
    }
  }

  // ---- epilogue: two 32-pixel passes through LDS, coalesced stores ----
  {
    const int n  = wave * 16 + (lane & 15);
    const int mb = (lane >> 4) * 8;
#pragma unroll
    for (int r = 0; r < 8; ++r) {
      sD[n][mb + r]      = acc0[r];
      sD[n][16 + mb + r] = acc1[r];
    }
  }
  __syncthreads();
  {
    float* outb = conv + (size_t)b * (COUT * HWSZ) + hw0 + lane;
#pragma unroll
    for (int i = 0; i < 16; ++i) {
      int o = i * 16 + wave;
      outb[(size_t)o * HWSZ] = sD[o][lane];
    }
  }
  __syncthreads();
  {
    const int n  = wave * 16 + (lane & 15);
    const int mb = (lane >> 4) * 8;
#pragma unroll
    for (int r = 0; r < 8; ++r) {
      sD[n][mb + r]      = acc2[r];
      sD[n][16 + mb + r] = acc3[r];
    }
  }
  __syncthreads();
  {
    float* outb = conv + (size_t)b * (COUT * HWSZ) + hw0 + 32 + lane;
#pragma unroll
    for (int i = 0; i < 16; ++i) {
      int o = i * 16 + wave;
      outb[(size_t)o * HWSZ] = sD[o][lane];
    }
  }
}

// ==========================================================================
// Kernel 2a: partial sums per (b,g) chunk (deterministic two-stage reduce)
// ==========================================================================
__global__ __launch_bounds__(256) void k_gn_partial(const float* __restrict__ conv,
                                                    float* __restrict__ part) {
  __shared__ float rs[256], rq[256];
  int bg = blockIdx.x / NCHUNK;
  int ch = blockIdx.x - bg * NCHUNK;
  const float* p = conv + (size_t)bg * GELEM + ch * 8192 + threadIdx.x;
  float s = 0.f, s2 = 0.f;
#pragma unroll
  for (int i = 0; i < 32; ++i) { float v = p[i * 256]; s += v; s2 += v * v; }
  rs[threadIdx.x] = s; rq[threadIdx.x] = s2;
  __syncthreads();
  for (int o = 128; o > 0; o >>= 1) {
    if (threadIdx.x < o) { rs[threadIdx.x] += rs[threadIdx.x + o];
                           rq[threadIdx.x] += rq[threadIdx.x + o]; }
    __syncthreads();
  }
  if (threadIdx.x == 0) { part[blockIdx.x * 2] = rs[0]; part[blockIdx.x * 2 + 1] = rq[0]; }
}

__global__ __launch_bounds__(64) void k_gn_final(const float* __restrict__ part,
                                                 float* __restrict__ stats) {
  __shared__ float rs[64], rq[64];
  int bg = blockIdx.x, t = threadIdx.x;
  float s = 0.f, s2 = 0.f;
  if (t < NCHUNK) { s = part[(bg * NCHUNK + t) * 2]; s2 = part[(bg * NCHUNK + t) * 2 + 1]; }
  rs[t] = s; rq[t] = s2;
  __syncthreads();
  for (int o = 32; o > 0; o >>= 1) {
    if (t < o) { rs[t] += rs[t + o]; rq[t] += rq[t + o]; }
    __syncthreads();
  }
  if (t == 0) {
    float inv = 1.f / (float)GELEM;
    float mu  = rs[0] * inv;
    float var = rq[0] * inv - mu * mu;
    stats[bg * 2]     = mu;
    stats[bg * 2 + 1] = rsqrtf(var + 1e-5f);
  }
}

// ==========================================================================
// Kernel 3: normalize + affine + ReLU (float4, coalesced)
// ==========================================================================
__global__ __launch_bounds__(256) void k_gn_apply(const float* __restrict__ conv,
                                                  const float* __restrict__ stats,
                                                  const float* __restrict__ gamma,
                                                  const float* __restrict__ beta,
                                                  float* __restrict__ out) {
  int idx = (blockIdx.x * 256 + threadIdx.x) * 4;
  int bo  = idx / HWSZ;
  int o   = bo & 255;
  int b   = bo >> 8;
  int bg  = b * GRP + (o >> 5);
  float mu = stats[bg * 2], rstd = stats[bg * 2 + 1];
  float gsc = gamma[o] * rstd;
  float gof = beta[o] - mu * gsc;
  float4 v = *(const float4*)(conv + idx);
  v.x = fmaxf(v.x * gsc + gof, 0.f);
  v.y = fmaxf(v.y * gsc + gof, 0.f);
  v.z = fmaxf(v.z * gsc + gof, 0.f);
  v.w = fmaxf(v.w * gsc + gof, 0.f);
  *(float4*)(out + idx) = v;
}

// ==========================================================================
extern "C" void kernel_launch(void* const* d_in, const int* in_sizes, int n_in,
                              void* d_out, int out_size, void* d_ws, size_t ws_size,
                              hipStream_t stream) {
  const float* x     = (const float*)d_in[0];
  const float* off   = (const float*)d_in[1];
  const float* msk   = (const float*)d_in[2];
  const float* wgt   = (const float*)d_in[3];
  const float* gamma = (const float*)d_in[4];
  const float* beta  = (const float*)d_in[5];
  float* out = (float*)d_out;

  char* ws = (char*)d_ws;
  float*        conv  = (float*)ws;                                   // 37,748,736 B
  unsigned int* WB    = (unsigned int*)(ws + 37748736);               //  1,179,648 B
  float*        part  = (float*)(ws + 37748736 + 1179648);            //      9,216 B
  float*        stats = part + (NB * GRP * NCHUNK) * 2;               //        256 B

  k_repack    <<<1152, 256, 0, stream>>>(wgt, WB);
  k_dcn_wmma  <<<NPOS / 64, 512, 0, stream>>>(x, off, msk, WB, conv);
  k_gn_partial<<<NB * GRP * NCHUNK, 256, 0, stream>>>(conv, part);
  k_gn_final  <<<NB * GRP, 64, 0, stream>>>(part, stats);
  k_gn_apply  <<<(NPOS * COUT) / 1024, 256, 0, stream>>>(conv, stats, gamma, beta, out);
}
